// HGNNPLUS_64029372449470
// MI455X (gfx1250) — matrix operand order repaired and checked by reading, not compiled, not run on hardware
//
#include <hip/hip_runtime.h>
#include <hip/hip_bf16.h>
#include <math.h>

// ---------------------------------------------------------------------------
// HGNN+ link predictor for MI455X (gfx1250, wave32).
//   - Dense GEMMs: v_wmma_f32_16x16x32_bf16, one wave computes a 16x64 output
//     strip (4 accumulators) so each A fragment (loads + bf16 cvt) feeds 4
//     WMMAs. W1/W2 (<=256KB) are fully L2/L0 resident.
//   - Segment-mean scatters: coalesced float4 gathers + global f32 atomics
//     (Xt=102MB and Yacc=20MB both fit the 192MB L2, atomics stay on-chip).
//   - Link readout: 1 wave32 per link, xor-shuffle dot-product reduction.
// ---------------------------------------------------------------------------

typedef __attribute__((ext_vector_type(16))) __bf16 v16bf;
typedef __attribute__((ext_vector_type(8)))  float  v8f;

#define NVERT 100000
#define MEDGE 20000
#define DIN   256
#define DH    256
#define DOUT  128

// ---------------- counts: segment_sum of ones (shared by both conv layers) --
__global__ void hgnn_count_kernel(const int* __restrict__ v_idx,
                                  const int* __restrict__ e_idx,
                                  float* __restrict__ cnt_v,
                                  float* __restrict__ cnt_e, int nnz) {
  int i = blockIdx.x * blockDim.x + threadIdx.x;
  if (i < nnz) {
    atomicAdd(&cnt_e[e_idx[i]], 1.0f);
    atomicAdd(&cnt_v[v_idx[i]], 1.0f);
  }
}

// ---------------- Out[rows,Cout] = A[rows,K] @ W[K,Cout] + bias ------------
// Block = 64 threads (2 waves). Wave w computes rows [blockIdx.x*16, +16) x
// cols [blockIdx.y*128 + w*64, +64) as four 16x16 WMMA tiles that share one
// A fragment per K-step. K-loop: 4x v_wmma_f32_16x16x32_bf16 per iteration.
__global__ void hgnn_gemm_bias_wmma(const float* __restrict__ A,
                                    const float* __restrict__ W,
                                    const float* __restrict__ bias,
                                    float* __restrict__ Out,
                                    int K, int Cout) {
  const int lane    = threadIdx.x & 31;
  const int wave    = threadIdx.x >> 5;                 // 0..1
  const int rowBase = blockIdx.x * 16;
  const int colBase = blockIdx.y * 128 + wave * 64;     // wave's 64-col strip
  const int halfSel = lane >> 4;                        // 0 | 1
  const int r       = lane & 15;

  const float* __restrict__ arow = A + (size_t)(rowBase + r) * K;

  v8f acc0 = {}, acc1 = {}, acc2 = {}, acc3 = {};
  for (int k0 = 0; k0 < K; k0 += 32) {
    // A fragment (16x32 bf16), contiguous K order matching B: lanes 0-15 row r
    // hold K k0+0..15; lanes 16-31 row r hold K k0+16..31.
    const int ka = k0 + halfSel * 16;
    v16bf afrag;
#pragma unroll
    for (int i = 0; i < 8; ++i) afrag[i]     = (__bf16)arow[ka + i];
#pragma unroll
    for (int i = 0; i < 8; ++i) afrag[8 + i] = (__bf16)arow[ka + 8 + i];

    // B fragments (32x16 bf16 each): lane holds one column; lanes 0-15 carry
    // K k0+0..15, lanes 16-31 carry K k0+16..31. Row-of-W loads are 16
    // consecutive floats across lanes -> coalesced, and W is cache-resident.
    const int kb = k0 + halfSel * 16;
    const float* __restrict__ wbase = W + (size_t)kb * Cout + colBase + r;
    v16bf b0, b1, b2, b3;
#pragma unroll
    for (int i = 0; i < 16; ++i) {
      const float* wr = wbase + (size_t)i * Cout;
      b0[i] = (__bf16)wr[0];
      b1[i] = (__bf16)wr[16];
      b2[i] = (__bf16)wr[32];
      b3[i] = (__bf16)wr[48];
    }

    acc0 = __builtin_amdgcn_wmma_f32_16x16x32_bf16(false, afrag, false, b0,
                                                   (short)0, acc0, false, false);
    acc1 = __builtin_amdgcn_wmma_f32_16x16x32_bf16(false, afrag, false, b1,
                                                   (short)0, acc1, false, false);
    acc2 = __builtin_amdgcn_wmma_f32_16x16x32_bf16(false, afrag, false, b2,
                                                   (short)0, acc2, false, false);
    acc3 = __builtin_amdgcn_wmma_f32_16x16x32_bf16(false, afrag, false, b3,
                                                   (short)0, acc3, false, false);
  }

  // C/D layout: VGPR g, lanes 0-15 -> (m=g, n=lane); lanes 16-31 -> (m=g+8).
  const float bb0 = bias[colBase + r];
  const float bb1 = bias[colBase + 16 + r];
  const float bb2 = bias[colBase + 32 + r];
  const float bb3 = bias[colBase + 48 + r];
  float* __restrict__ obase =
      Out + (size_t)(rowBase + halfSel * 8) * Cout + colBase + r;
#pragma unroll
  for (int g = 0; g < 8; ++g) {
    float* orow = obase + (size_t)g * Cout;
    orow[0]  = acc0[g] + bb0;
    orow[16] = acc1[g] + bb1;
    orow[32] = acc2[g] + bb2;
    orow[48] = acc3[g] + bb3;
  }
}

// ---------------- scatter-add: acc[dst_idx[p]] += src[src_idx[p]] ----------
// Thread = (pair, 4-channel chunk); consecutive threads hit consecutive
// channels of one pair -> coalesced gather + coalesced atomic stream.
__global__ void hgnn_scatter_add(const float* __restrict__ src,
                                 const int* __restrict__ src_idx,
                                 const int* __restrict__ dst_idx,
                                 float* __restrict__ acc,
                                 int nnz, int log2chunks /* C/4 = 1<<log2chunks */) {
  const long long t     = (long long)blockIdx.x * blockDim.x + threadIdx.x;
  const long long total = (long long)nnz << log2chunks;
  if (t >= total) return;
  const int p  = (int)(t >> log2chunks);
  const int c4 = ((int)t & ((1 << log2chunks) - 1)) << 2;
  const int C  = 4 << log2chunks;

  const float4 v = *(const float4*)(src + (size_t)src_idx[p] * C + c4);
  float* d = acc + (size_t)dst_idx[p] * C + c4;
  atomicAdd(d + 0, v.x);
  atomicAdd(d + 1, v.y);
  atomicAdd(d + 2, v.z);
  atomicAdd(d + 3, v.w);
}

// ---------------- in-place mean normalize (+ optional ReLU) ----------------
__global__ void hgnn_normalize(float* __restrict__ buf,
                               const float* __restrict__ cnt,
                               long long total, int log2C, int do_relu) {
  const long long t = (long long)blockIdx.x * blockDim.x + threadIdx.x;
  if (t >= total) return;
  const int row = (int)(t >> log2C);
  float v = buf[t] / fmaxf(cnt[row], 1.0f);
  if (do_relu) v = fmaxf(v, 0.0f);
  buf[t] = v;
}

// ---------------- link readout: sigmoid(mean(h2[link]) @ fc_w + fc_b) ------
// One wave32 per link; 4 channels/lane (DOUT=128); xor-shuffle reduction.
__global__ void hgnn_link_kernel(const float* __restrict__ h2,
                                 const int* __restrict__ link,
                                 const float* __restrict__ fc_w,
                                 const float* __restrict__ fc_b,
                                 float* __restrict__ out, int L) {
  const int w    = (int)((blockIdx.x * (long long)blockDim.x + threadIdx.x) >> 5);
  const int lane = threadIdx.x & 31;
  if (w >= L) return;
  const int va = link[2 * w + 0];
  const int vb = link[2 * w + 1];

  float partial = 0.0f;
#pragma unroll
  for (int i = 0; i < 4; ++i) {
    const int c = lane + i * 32;
    const float e = 0.5f * (h2[(size_t)va * DOUT + c] + h2[(size_t)vb * DOUT + c]);
    partial += e * fc_w[c];
  }
#pragma unroll
  for (int off = 16; off > 0; off >>= 1)
    partial += __shfl_xor(partial, off, 32);

  if (lane == 0) {
    const float z = partial + fc_b[0];
    out[w] = 1.0f / (1.0f + __expf(-z));
  }
}

// ---------------------------------------------------------------------------
extern "C" void kernel_launch(void* const* d_in, const int* in_sizes, int n_in,
                              void* d_out, int out_size, void* d_ws, size_t ws_size,
                              hipStream_t stream) {
  const float* X    = (const float*)d_in[0];   // [N, 256]
  const float* W1   = (const float*)d_in[1];   // [256, 256]
  const float* b1   = (const float*)d_in[2];   // [256]
  const float* W2   = (const float*)d_in[3];   // [256, 128]
  const float* b2   = (const float*)d_in[4];   // [128]
  const float* fc_w = (const float*)d_in[5];   // [128, 1]
  const float* fc_b = (const float*)d_in[6];   // [1]
  const int*   vidx = (const int*)d_in[7];     // [NNZ]
  const int*   eidx = (const int*)d_in[8];     // [NNZ]
  const int*   link = (const int*)d_in[9];     // [L, 2]

  const int nnz = in_sizes[7];
  const int L   = out_size;
  float* out = (float*)d_out;

  // ---- workspace carve (floats) ----
  float* ws    = (float*)d_ws;
  float* cnt_e = ws;                                  // [M]
  float* cnt_v = cnt_e + MEDGE;                       // [N]
  float* theta = cnt_v + NVERT;                       // [N,256] (Xt, reused as Xt2)
  float* eacc  = theta + (size_t)NVERT * DH;          // [M,256] (reused [M,128])
  float* h1    = eacc  + (size_t)MEDGE * DH;          // [N,256]
  float* h2    = h1    + (size_t)NVERT * DH;          // [N,128]

  const int TB = 256;
  const dim3 blk256(TB);

  // ---- 0) segment counts (shared by both layers) ----
  hipMemsetAsync(cnt_e, 0, sizeof(float) * MEDGE, stream);
  hipMemsetAsync(cnt_v, 0, sizeof(float) * NVERT, stream);
  hgnn_count_kernel<<<(nnz + TB - 1) / TB, blk256, 0, stream>>>(vidx, eidx, cnt_v, cnt_e, nnz);

  // ---- conv layer 1: theta = X@W1+b1; v2e mean; e2v mean; relu -> h1 ----
  hgnn_gemm_bias_wmma<<<dim3(NVERT / 16, DH / 128), dim3(64), 0, stream>>>(
      X, W1, b1, theta, DIN, DH);

  hipMemsetAsync(eacc, 0, sizeof(float) * (size_t)MEDGE * DH, stream);
  {
    long long total = (long long)nnz * (DH / 4);
    hgnn_scatter_add<<<(unsigned)((total + TB - 1) / TB), blk256, 0, stream>>>(
        theta, vidx, eidx, eacc, nnz, 6 /* 64 chunks */);
  }
  {
    long long total = (long long)MEDGE * DH;
    hgnn_normalize<<<(unsigned)((total + TB - 1) / TB), blk256, 0, stream>>>(
        eacc, cnt_e, total, 8, 0);
  }

  hipMemsetAsync(h1, 0, sizeof(float) * (size_t)NVERT * DH, stream);
  {
    long long total = (long long)nnz * (DH / 4);
    hgnn_scatter_add<<<(unsigned)((total + TB - 1) / TB), blk256, 0, stream>>>(
        eacc, eidx, vidx, h1, nnz, 6);
  }
  {
    long long total = (long long)NVERT * DH;
    hgnn_normalize<<<(unsigned)((total + TB - 1) / TB), blk256, 0, stream>>>(
        h1, cnt_v, total, 8, 1 /* relu */);
  }

  // ---- conv layer 2: theta2 = h1@W2+b2 (reuse theta); v2e; e2v; relu -> h2 ----
  hgnn_gemm_bias_wmma<<<dim3(NVERT / 16, DOUT / 128), dim3(64), 0, stream>>>(
      h1, W2, b2, theta, DH, DOUT);

  hipMemsetAsync(eacc, 0, sizeof(float) * (size_t)MEDGE * DOUT, stream);
  {
    long long total = (long long)nnz * (DOUT / 4);
    hgnn_scatter_add<<<(unsigned)((total + TB - 1) / TB), blk256, 0, stream>>>(
        theta, vidx, eidx, eacc, nnz, 5 /* 32 chunks */);
  }
  {
    long long total = (long long)MEDGE * DOUT;
    hgnn_normalize<<<(unsigned)((total + TB - 1) / TB), blk256, 0, stream>>>(
        eacc, cnt_e, total, 7, 0);
  }

  hipMemsetAsync(h2, 0, sizeof(float) * (size_t)NVERT * DOUT, stream);
  {
    long long total = (long long)nnz * (DOUT / 4);
    hgnn_scatter_add<<<(unsigned)((total + TB - 1) / TB), blk256, 0, stream>>>(
        eacc, eidx, vidx, h2, nnz, 5);
  }
  {
    long long total = (long long)NVERT * DOUT;
    hgnn_normalize<<<(unsigned)((total + TB - 1) / TB), blk256, 0, stream>>>(
        h2, cnt_v, total, 7, 1 /* relu */);
  }

  // ---- link readout ----
  {
    const int wavesPerBlock = TB / 32;
    const int nblocks = (L + wavesPerBlock - 1) / wavesPerBlock;
    hgnn_link_kernel<<<nblocks, blk256, 0, stream>>>(h2, link, fc_w, fc_b, out, L);
  }
}
